// DCNConv_74019466379809
// MI455X (gfx1250) — matrix operand order repaired and checked by reading, not compiled, not run on hardware
//
#include <hip/hip_runtime.h>
#include <hip/hip_bf16.h>

#define BATCH 2
#define CCH   64
#define HH    384
#define WW    384
#define GG    4
#define GCH   16
#define OMD   112          // om logical width
#define OMP   116          // om padded row stride (floats) -> conflict-free banks
#define ASTR  72           // A/B staging row stride (bf16 elems), 144B = 9*16B
#define HWPIX (HH*WW)

typedef __attribute__((ext_vector_type(16))) __bf16 v16bf;
typedef __attribute__((ext_vector_type(8)))  __bf16 v8bf;
typedef __attribute__((ext_vector_type(8)))  float  v8f;
typedef __attribute__((ext_vector_type(4)))  float  v4f;

// ---------------------------------------------------------------------------
// WMMA fragment loaders, wave32, per CDNA5 ISA 7.12.2 layouts.
// A staged row-major [16][ASTR] bf16: lane m=lane&15; lanes<16 hold
// K {0..7,16..23}, lanes>=16 hold K {8..15,24..31} of the kchunk.
// ---------------------------------------------------------------------------
__device__ __forceinline__ v16bf load_fragA(const __bf16* a, int kchunk, int lane) {
    int m     = lane & 15;
    int khalf = (lane >> 4) << 3;            // 0 or 8
    const __bf16* row = a + m * ASTR + kchunk + khalf;
    v8bf lo = *(const v8bf*)(row);           // K = khalf .. khalf+7
    v8bf hi = *(const v8bf*)(row + 16);      // K = 16+khalf .. +7
    v16bf r;
#pragma unroll
    for (int i = 0; i < 8; ++i) { r[i] = lo[i]; r[8 + i] = hi[i]; }
    return r;
}

// B: 32x16 (KxN). Weights staged transposed: wt[n*ASTR + k] (K contiguous).
// lanes<16 hold K {0..7,16..23} of column n=lane, lanes>=16 hold K {8..15,24..31}.
__device__ __forceinline__ v16bf load_fragB(const __bf16* wt, int ntile, int kchunk, int lane) {
    int n     = lane & 15;
    int khalf = (lane >> 4) << 3;            // 0 or 8
    const __bf16* col = wt + (ntile * 16 + n) * ASTR + kchunk + khalf;
    v8bf lo = *(const v8bf*)(col);
    v8bf hi = *(const v8bf*)(col + 16);
    v16bf r;
#pragma unroll
    for (int i = 0; i < 8; ++i) { r[i] = lo[i]; r[8 + i] = hi[i]; }
    return r;
}

// ---------------------------------------------------------------------------
// Kernel 1: value = x @ Wv + bv   -> workspace, layout (B,H,W,64) fp32
// ---------------------------------------------------------------------------
__global__ __launch_bounds__(128) void dcn_value_kernel(
    const float* __restrict__ inp, const float* __restrict__ Wv,
    const float* __restrict__ bv, float* __restrict__ value) {

    __shared__ __attribute__((aligned(16))) __bf16 sWt[64 * ASTR];   // Wv^T
    __shared__ float  sBias[64];
    __shared__ __attribute__((aligned(16))) __bf16 sA[4][16 * ASTR];
    __shared__ __attribute__((aligned(16))) float  sOut[64 * 64];    // block tile

    int t = threadIdx.x;
    for (int i = t; i < 64 * 64; i += 128) {
        int k = i & 63, n = i >> 6;
        sWt[n * ASTR + k] = (__bf16)Wv[k * 64 + n];
    }
    if (t < 64) sBias[t] = bv[t];
    __syncthreads();

    int wave = t >> 5, lane = t & 31;
    long tileBase = (long)blockIdx.x * 64;
    long tilePix  = tileBase + wave * 16;
    int b   = (int)(tilePix / HWPIX);
    int rem = (int)(tilePix % HWPIX);
    int y   = rem / WW, xw = rem % WW;

    // Stage A: 16 pixels x 64 input channels (strided NCHW reads -> bf16 LDS)
    {
        int p = lane & 15;
        int cbase = (lane >> 4) * 32;
        const float* src = inp + (((long)b * CCH + cbase) * HH + y) * WW + (xw + p);
        __bf16* dst = &sA[wave][p * ASTR + cbase];
#pragma unroll
        for (int j = 0; j < 32; ++j)
            dst[j] = (__bf16)src[(long)j * HWPIX];
    }
    __syncthreads();

    v16bf a0 = load_fragA(sA[wave], 0,  lane);
    v16bf a1 = load_fragA(sA[wave], 32, lane);
    int n  = lane & 15;
    int hi = lane >> 4;
#pragma unroll
    for (int nt = 0; nt < 4; ++nt) {
        float bias = sBias[nt * 16 + n];
        v8f c;
#pragma unroll
        for (int r = 0; r < 8; ++r) c[r] = bias;       // bias folded into C
        v16bf b0 = load_fragB(sWt, nt, 0,  lane);
        v16bf b1 = load_fragB(sWt, nt, 32, lane);
        c = __builtin_amdgcn_wmma_f32_16x16x32_bf16(false, a0, false, b0, (short)0, c, false, false);
        c = __builtin_amdgcn_wmma_f32_16x16x32_bf16(false, a1, false, b1, (short)0, c, false, false);
#pragma unroll
        for (int r = 0; r < 8; ++r) {
            int m = r + (hi << 3);
            sOut[(wave * 16 + m) * 64 + nt * 16 + n] = c[r];   // pixel-major
        }
    }
    __syncthreads();

    // Fully coalesced b128 stores: block tile is one contiguous 16KB region.
    float* dst = value + tileBase * 64;
#pragma unroll
    for (int it = 0; it < 8; ++it) {
        int idx = (it * 128 + t) * 4;
        *(v4f*)(dst + idx) = *(const v4f*)(sOut + idx);
    }
}

// ---------------------------------------------------------------------------
// Kernel 2 (fused): depthwise 3x3 -> om GEMM -> deformable bilinear sampling
//                   -> Wo GEMM -> NCHW store
// ---------------------------------------------------------------------------
__global__ __launch_bounds__(128) void dcn_fused_kernel(
    const float* __restrict__ inp,  const float* __restrict__ Wdw,
    const float* __restrict__ bdw,  const float* __restrict__ Wom,
    const float* __restrict__ bom,  const float* __restrict__ Wo,
    const float* __restrict__ value, float* __restrict__ out) {

    __shared__ __attribute__((aligned(16))) __bf16 sWomT[OMD * ASTR]; // 16128 B
    __shared__ float  sBom[OMD];
    __shared__ float  sWdw[64 * 9];
    __shared__ float  sBdw[64];
    __shared__ __attribute__((aligned(16))) __bf16 sWoT[64 * ASTR];   // 9216 B
    __shared__ __attribute__((aligned(16))) __bf16 sA[4][16 * ASTR];  // dw, then out_pre
    __shared__ __attribute__((aligned(16))) float  sOm[4][16 * OMP];  // om; reused as out tile

    int t = threadIdx.x;
    long tileBase = (long)blockIdx.x * 64;
    int b   = (int)(tileBase / HWPIX);
    int rem = (int)(tileBase % HWPIX);
    int y   = rem / WW, xb = rem % WW;

    // Prefetch the depthwise 3x3 input rows for this block (global_prefetch_b8)
    for (int i = t; i < 64 * 3 * 2; i += 128) {
        int c  = i >> 2;                  // wait: 64*3*2/4... use explicit decode
        int r3 = (i >> 1) % 3;
        int half = i & 1;
        c = i / 6;
        int yy = y + r3 - 1;
        if (yy >= 0 && yy < HH)
            __builtin_prefetch(inp + (((long)b * CCH + c) * HH + yy) * WW + xb + half * 32, 0, 3);
    }

    for (int i = t; i < OMD * 64; i += 128) {
        int k = i & 63, n = i >> 6;
        sWomT[n * ASTR + k] = (__bf16)Wom[k * OMD + n];
    }
    for (int i = t; i < 64 * 64; i += 128) {
        int k = i & 63, n = i >> 6;
        sWoT[n * ASTR + k] = (__bf16)Wo[k * 64 + n];
    }
    for (int i = t; i < 64 * 9; i += 128) sWdw[i] = Wdw[i];
    if (t < OMD) sBom[t] = bom[t];
    if (t < 64)  sBdw[t] = bdw[t];
    __syncthreads();

    int wave = t >> 5, lane = t & 31;
    int xw = xb + wave * 16;

    int p  = lane & 15;
    int hi = lane >> 4;            // 0/1: lane half
    int gx = xw + p;               // global x of this lane's pixel

    // ---- Phase 1: depthwise 3x3 conv + bias -> bf16 A tile -------------------
    {
        int cbase = hi * 32;
#pragma unroll 4
        for (int j = 0; j < 32; ++j) {
            int c = cbase + j;
            const float* base = inp + ((long)b * CCH + c) * HWPIX;
            float s = sBdw[c];
#pragma unroll
            for (int ky = 0; ky < 3; ++ky) {
                int yy = y + ky - 1;
                if (yy < 0 || yy >= HH) continue;
#pragma unroll
                for (int kx = 0; kx < 3; ++kx) {
                    int xx = gx + kx - 1;
                    if (xx < 0 || xx >= WW) continue;
                    s += base[(long)yy * WW + xx] * sWdw[c * 9 + ky * 3 + kx];
                }
            }
            sA[wave][p * ASTR + c] = (__bf16)s;
        }
    }
    __syncthreads();

    // ---- Phase 2: om = dw @ Wom + bom  (M=16, N=112, K=64) -------------------
    {
        v16bf a0 = load_fragA(sA[wave], 0,  lane);
        v16bf a1 = load_fragA(sA[wave], 32, lane);
        int n = lane & 15;
#pragma unroll
        for (int nt = 0; nt < 7; ++nt) {
            float bias = sBom[nt * 16 + n];
            v8f c;
#pragma unroll
            for (int r = 0; r < 8; ++r) c[r] = bias;
            v16bf b0 = load_fragB(sWomT, nt, 0,  lane);
            v16bf b1 = load_fragB(sWomT, nt, 32, lane);
            c = __builtin_amdgcn_wmma_f32_16x16x32_bf16(false, a0, false, b0, (short)0, c, false, false);
            c = __builtin_amdgcn_wmma_f32_16x16x32_bf16(false, a1, false, b1, (short)0, c, false, false);
#pragma unroll
            for (int r = 0; r < 8; ++r)
                sOm[wave][(r + (hi << 3)) * OMP + nt * 16 + n] = c[r];
        }
    }
    __syncthreads();

    // ---- Phase 3: deformable bilinear sampling -------------------------------
    // lane covers pixel p, channels [hi*32, hi*32+32) == groups {2*hi, 2*hi+1}
    float acc[32];
#pragma unroll
    for (int i = 0; i < 32; ++i) acc[i] = 0.f;

    {
        const float* om = &sOm[wave][p * OMP];
#pragma unroll
        for (int gl = 0; gl < 2; ++gl) {
            int g = (hi << 1) + gl;
            const float* omg = om + g * 27;   // 18 offsets + 9 masks
            for (int k = 0; k < 9; ++k) {
                float dx  = omg[2 * k];
                float dy  = omg[2 * k + 1];
                float msk = omg[18 + k];
                float px = (float)gx + (float)(k % 3 - 1) + dx;
                float py = (float)y  + (float)(k / 3 - 1) + dy;
                float fx0 = floorf(px), fy0 = floorf(py);
                float wx1 = px - fx0,  wy1 = py - fy0;
                int x0i = (int)fx0, y0i = (int)fy0;
#pragma unroll
                for (int cy = 0; cy < 2; ++cy) {
                    int   yi = y0i + cy;
                    float wy = cy ? wy1 : 1.f - wy1;
                    bool  yok = (yi >= 0) && (yi < HH);
                    int   yc  = min(max(yi, 0), HH - 1);
#pragma unroll
                    for (int cx = 0; cx < 2; ++cx) {
                        int   xi = x0i + cx;
                        float wx = cx ? wx1 : 1.f - wx1;
                        bool  ok = yok && (xi >= 0) && (xi < WW);
                        int   xc = min(max(xi, 0), WW - 1);
                        float w  = msk * wy * wx * (ok ? 1.f : 0.f);
                        const float* vp =
                            value + (((long)b * HH + yc) * WW + xc) * 64 + g * GCH;
#pragma unroll
                        for (int q = 0; q < 4; ++q) {
                            v4f v = *(const v4f*)(vp + 4 * q);
                            acc[gl * 16 + 4 * q + 0] += w * v[0];
                            acc[gl * 16 + 4 * q + 1] += w * v[1];
                            acc[gl * 16 + 4 * q + 2] += w * v[2];
                            acc[gl * 16 + 4 * q + 3] += w * v[3];
                        }
                    }
                }
            }
        }
    }
    __syncthreads();   // om reads done; safe to overwrite sA / sOm

    // Stage out_pre as bf16 A tile
    {
        int cbase = hi * 32;
#pragma unroll
        for (int i = 0; i < 32; ++i)
            sA[wave][p * ASTR + cbase + i] = (__bf16)acc[i];
    }
    __syncthreads();

    // ---- Phase 4: out = out_pre @ Wo  (M=16, N=64, K=64) ---------------------
    // D tiles go to block-wide channel-major LDS buffer (reuse sOm storage).
    float* trB = &sOm[0][0];               // [64 ch][64 px], 16KB of the 29KB
    {
        v16bf a0 = load_fragA(sA[wave], 0,  lane);
        v16bf a1 = load_fragA(sA[wave], 32, lane);
        int n = lane & 15;
#pragma unroll
        for (int nt = 0; nt < 4; ++nt) {
            v8f c;
#pragma unroll
            for (int r = 0; r < 8; ++r) c[r] = 0.f;
            v16bf b0 = load_fragB(sWoT, nt, 0,  lane);
            v16bf b1 = load_fragB(sWoT, nt, 32, lane);
            c = __builtin_amdgcn_wmma_f32_16x16x32_bf16(false, a0, false, b0, (short)0, c, false, false);
            c = __builtin_amdgcn_wmma_f32_16x16x32_bf16(false, a1, false, b1, (short)0, c, false, false);
#pragma unroll
            for (int r = 0; r < 8; ++r) {
                int m = r + (hi << 3);
                trB[(nt * 16 + n) * 64 + wave * 16 + m] = c[r];
            }
        }
    }
    __syncthreads();

    // ---- Phase 5: coalesced NCHW store (b128, 256B contiguous per channel) ---
#pragma unroll
    for (int it = 0; it < 8; ++it) {
        int idx = (it * 128 + t) * 4;      // flat over [ch][px], quads stay in-row
        int ch  = idx >> 6;
        int px  = idx & 63;
        v4f v = *(const v4f*)(trB + idx);
        *(v4f*)(out + (((long)b * CCH + ch) * HH + y) * WW + xb + px) = v;
    }
}

// ---------------------------------------------------------------------------
extern "C" void kernel_launch(void* const* d_in, const int* in_sizes, int n_in,
                              void* d_out, int out_size, void* d_ws, size_t ws_size,
                              hipStream_t stream) {
    const float* inp = (const float*)d_in[0];
    const float* Wv  = (const float*)d_in[1];
    const float* bv  = (const float*)d_in[2];
    const float* Wdw = (const float*)d_in[3];
    const float* bdw = (const float*)d_in[4];
    const float* Wom = (const float*)d_in[5];
    const float* bom = (const float*)d_in[6];
    const float* Wo  = (const float*)d_in[7];
    float* out   = (float*)d_out;
    float* value = (float*)d_ws;                 // B*H*W*64 fp32 = 75.5 MB

    const int nBlocks = (BATCH * HWPIX) / 64;    // 4608 blocks, 4 waves x 16 px
    hipLaunchKernelGGL(dcn_value_kernel, dim3(nBlocks), dim3(128), 0, stream,
                       inp, Wv, bv, value);
    hipLaunchKernelGGL(dcn_fused_kernel, dim3(nBlocks), dim3(128), 0, stream,
                       inp, Wdw, bdw, Wom, bom, Wo, value, out);
}
